// Qwen3MoeAfdDecoderLayer_6073083756732
// MI455X (gfx1250) — compile-verified, matrix operand order checked by reading, not run on hardware
//
#include <hip/hip_runtime.h>

// Problem sizes (fixed by the reference)
#define T_TOK 1024
#define HDIM  2048
#define NHEAD 16
#define HD    128
#define NE    32
#define TOPK  8
#define FDIM  768

// GEMM tiling: 128x64 block tile, K-step 32, 8 waves each owning a 32x32
// subtile (4 x v_wmma_f32_16x16x32_f16 per wave per k-step), double-buffered LDS.
#define BM 128
#define BN 64
#define BK 32
#define LDT 48   // LDS row stride in halves (96B: keeps all b128 frag ops 16B-aligned)

typedef __attribute__((ext_vector_type(16))) _Float16 v16h;
typedef __attribute__((ext_vector_type(8)))  _Float16 v8h;
typedef __attribute__((ext_vector_type(8)))  float    v8f;

// ---------------------------------------------------------------------------
// WMMA fragment load from an LDS tile stored K-contiguous per row (A tile) or
// per column (B^T tile). ISA 7.12.2 16-bit layout: lane&15 = row (A) / col (B),
// halves 0..7 -> K = koff..koff+7, halves 8..15 -> K = 16+koff..16+koff+7,
// koff = ((lane>>4)&1)*8.  Two aligned ds_load_b128 per fragment.
// ---------------------------------------------------------------------------
__device__ __forceinline__ v16h frag_ld(const _Float16* base, int lane) {
    int row  = lane & 15;
    int koff = ((lane >> 4) & 1) << 3;
    const _Float16* p = base + row * LDT + koff;
    v8h lo = *(const v8h*)(p);
    v8h hi = *(const v8h*)(p + 16);
    v16h r;
#pragma unroll
    for (int i = 0; i < 8; ++i) { r[i] = lo[i]; r[8 + i] = hi[i]; }
    return r;
}

__device__ __forceinline__ v8f wmma(v16h a, v16h b, v8f c) {
    return __builtin_amdgcn_wmma_f32_16x16x32_f16(false, a, false, b, (short)0, c,
                                                  false, false);
}

__device__ __forceinline__ float siluf(float x) {
    return x / (1.0f + __expf(-x));
}

// ---------------------------------------------------------------------------
// RMSNorm (fp32 in) -> f16 out
// ---------------------------------------------------------------------------
__global__ void rmsnorm_kernel(const float* __restrict__ x, const float* __restrict__ g,
                               _Float16* __restrict__ out) {
    __shared__ float red[256];
    int row = blockIdx.x, tid = threadIdx.x;
    const float* xr = x + (size_t)row * HDIM;
    float s = 0.f;
    for (int i = tid; i < HDIM; i += 256) { float v = xr[i]; s += v * v; }
    red[tid] = s; __syncthreads();
    for (int k = 128; k > 0; k >>= 1) { if (tid < k) red[tid] += red[tid + k]; __syncthreads(); }
    float rms = rsqrtf(red[0] / (float)HDIM + 1e-6f);
    _Float16* orow = out + (size_t)row * HDIM;
    for (int i = tid; i < HDIM; i += 256) orow[i] = (_Float16)(xr[i] * rms * g[i]);
}

// ---------------------------------------------------------------------------
// Generic GEMM: C[M,N] = A_f16[M,K] @ B_f32[K,N]
//   OUTMODE 0: store f16
//   OUTMODE 1: store fp32 = C + Res (residual), duplicated to Cf and Cf2
// A loader: thread owns 16 contiguous halves of one row (2 x global b128).
// B loader: thread owns one column n and 8 strided k-floats (coalesced dwords
//           across lanes), packed to f16, one aligned ds_store_b128.
// ---------------------------------------------------------------------------
template <int OUTMODE>
__global__ __launch_bounds__(256) void gemm_aw_kernel(
    const _Float16* __restrict__ A, const float* __restrict__ B,
    int K, int N,
    _Float16* __restrict__ Ch, float* __restrict__ Cf,
    const float* __restrict__ Res, float* __restrict__ Cf2) {
    __shared__ __align__(16) _Float16 lA[2][BM * LDT];
    __shared__ __align__(16) _Float16 lB[2][BN * LDT];
    int tid = threadIdx.x, lane = tid & 31, wave = tid >> 5;
    int m0 = blockIdx.y * BM, n0 = blockIdx.x * BN;
    int wm = (wave & 3) * 32, wn = (wave >> 2) * 32;
    int am = tid >> 1, ak = (tid & 1) * 16;
    int bn = tid & 63, bk = (tid >> 6) * 8;
    const _Float16* Ap = A + (size_t)(m0 + am) * K + ak;
    const float* Bp = B + n0 + bn;

    v8h a0, a1; float bs[8];
    auto fetch = [&](int k0) {
        a0 = *(const v8h*)(Ap + k0);
        a1 = *(const v8h*)(Ap + k0 + 8);
        __builtin_prefetch(&Bp[(size_t)(k0 + BK + bk) * N], 0, 1);
#pragma unroll
        for (int i = 0; i < 8; ++i) bs[i] = Bp[(size_t)(k0 + bk + i) * N];
    };
    auto stage = [&](int buf) {
        *(v8h*)&lA[buf][am * LDT + ak] = a0;
        *(v8h*)&lA[buf][am * LDT + ak + 8] = a1;
        v8h hb;
#pragma unroll
        for (int i = 0; i < 8; ++i) hb[i] = (_Float16)bs[i];
        *(v8h*)&lB[buf][bn * LDT + bk] = hb;
    };

    fetch(0); stage(0);
    v8f c00 = {}, c01 = {}, c10 = {}, c11 = {};
    int buf = 0;
    for (int k0 = 0; k0 < K; k0 += BK, buf ^= 1) {
        __syncthreads();
        bool nxt = (k0 + BK) < K;
        if (nxt) fetch(k0 + BK);
        v16h af0 = frag_ld(&lA[buf][wm * LDT], lane);
        v16h af1 = frag_ld(&lA[buf][(wm + 16) * LDT], lane);
        v16h bf0 = frag_ld(&lB[buf][wn * LDT], lane);
        v16h bf1 = frag_ld(&lB[buf][(wn + 16) * LDT], lane);
        c00 = wmma(af0, bf0, c00); c01 = wmma(af0, bf1, c01);
        c10 = wmma(af1, bf0, c10); c11 = wmma(af1, bf1, c11);
        if (nxt) stage(buf ^ 1);
    }
    int col = lane & 15, rb = (lane >> 4) * 8;
#pragma unroll
    for (int r = 0; r < 8; ++r) {
        int mmA = m0 + wm + rb + r, mmB = mmA + 16;
        size_t i00 = (size_t)mmA * N + n0 + wn + col, i01 = i00 + 16;
        size_t i10 = (size_t)mmB * N + n0 + wn + col, i11 = i10 + 16;
        if constexpr (OUTMODE == 0) {
            Ch[i00] = (_Float16)c00[r]; Ch[i01] = (_Float16)c01[r];
            Ch[i10] = (_Float16)c10[r]; Ch[i11] = (_Float16)c11[r];
        } else {
            float v00 = c00[r] + Res[i00], v01 = c01[r] + Res[i01];
            float v10 = c10[r] + Res[i10], v11 = c11[r] + Res[i11];
            Cf[i00] = v00; Cf2[i00] = v00; Cf[i01] = v01; Cf2[i01] = v01;
            Cf[i10] = v10; Cf2[i10] = v10; Cf[i11] = v11; Cf2[i11] = v11;
        }
    }
}

// ---------------------------------------------------------------------------
// Attention scores for one head: S[q,k] = Q_h[q,:] . K_h[k,:]  (raw dot, fp32)
// K rows are already B^T form -> both loaders are straight b128 row copies.
// ---------------------------------------------------------------------------
__global__ __launch_bounds__(256) void attn_score_kernel(
    const _Float16* __restrict__ Q, const _Float16* __restrict__ Km,
    float* __restrict__ S, int head) {
    int m0 = blockIdx.y * BM, n0 = blockIdx.x * BN;
    if (n0 >= m0 + BM) return;  // fully masked tile
    __shared__ __align__(16) _Float16 lA[2][BM * LDT];
    __shared__ __align__(16) _Float16 lB[2][BN * LDT];
    int tid = threadIdx.x, lane = tid & 31, wave = tid >> 5;
    int wm = (wave & 3) * 32, wn = (wave >> 2) * 32;
    int am = tid >> 1, ak = (tid & 1) * 16;
    int bnn = tid >> 2, bkk = (tid & 3) * 8;
    int cbase = head * HD;
    const _Float16* Ap = Q + (size_t)(m0 + am) * HDIM + cbase + ak;
    const _Float16* Bp = Km + (size_t)(n0 + bnn) * HDIM + cbase + bkk;

    v8h a0, a1, b0;
    auto fetch = [&](int k0) {
        a0 = *(const v8h*)(Ap + k0);
        a1 = *(const v8h*)(Ap + k0 + 8);
        b0 = *(const v8h*)(Bp + k0);
    };
    auto stage = [&](int buf) {
        *(v8h*)&lA[buf][am * LDT + ak] = a0;
        *(v8h*)&lA[buf][am * LDT + ak + 8] = a1;
        *(v8h*)&lB[buf][bnn * LDT + bkk] = b0;
    };

    fetch(0); stage(0);
    v8f c00 = {}, c01 = {}, c10 = {}, c11 = {};
    int buf = 0;
    for (int k0 = 0; k0 < HD; k0 += BK, buf ^= 1) {
        __syncthreads();
        bool nxt = (k0 + BK) < HD;
        if (nxt) fetch(k0 + BK);
        v16h af0 = frag_ld(&lA[buf][wm * LDT], lane);
        v16h af1 = frag_ld(&lA[buf][(wm + 16) * LDT], lane);
        v16h bf0 = frag_ld(&lB[buf][wn * LDT], lane);
        v16h bf1 = frag_ld(&lB[buf][(wn + 16) * LDT], lane);
        c00 = wmma(af0, bf0, c00); c01 = wmma(af0, bf1, c01);
        c10 = wmma(af1, bf0, c10); c11 = wmma(af1, bf1, c11);
        if (nxt) stage(buf ^ 1);
    }
    int col = lane & 15, rb = (lane >> 4) * 8;
#pragma unroll
    for (int r = 0; r < 8; ++r) {
        int mmA = m0 + wm + rb + r, mmB = mmA + 16;
        S[(size_t)mmA * T_TOK + n0 + wn + col]      = c00[r];
        S[(size_t)mmA * T_TOK + n0 + wn + 16 + col] = c01[r];
        S[(size_t)mmB * T_TOK + n0 + wn + col]      = c10[r];
        S[(size_t)mmB * T_TOK + n0 + wn + 16 + col] = c11[r];
    }
}

// ---------------------------------------------------------------------------
// Causal softmax over one score row -> P (f16, zero beyond the diagonal)
// ---------------------------------------------------------------------------
__global__ void softmax_kernel(const float* __restrict__ S, _Float16* __restrict__ P) {
    __shared__ float red[256];
    int row = blockIdx.x, tid = threadIdx.x;
    const float scale = 0.08838834764831845f;  // 1/sqrt(128)
    int valid = row + 1;
    const float* srow = S + (size_t)row * T_TOK;
    float m = -3.4e38f;
    for (int i = tid; i < valid; i += 256) m = fmaxf(m, srow[i] * scale);
    red[tid] = m; __syncthreads();
    for (int k = 128; k > 0; k >>= 1) { if (tid < k) red[tid] = fmaxf(red[tid], red[tid + k]); __syncthreads(); }
    m = red[0]; __syncthreads();
    float s = 0.f;
    for (int i = tid; i < valid; i += 256) s += __expf(srow[i] * scale - m);
    red[tid] = s; __syncthreads();
    for (int k = 128; k > 0; k >>= 1) { if (tid < k) red[tid] += red[tid + k]; __syncthreads(); }
    float inv = 1.0f / red[0];
    _Float16* prow = P + (size_t)row * T_TOK;
    for (int i = tid; i < T_TOK; i += 256)
        prow[i] = (i < valid) ? (_Float16)(__expf(srow[i] * scale - m) * inv) : (_Float16)0.0f;
}

// ---------------------------------------------------------------------------
// O_h = P @ V_h for one head (k-loop clipped to the causal diagonal).
// V loader: thread owns one d-column, 8 strided k-halves, one ds_store_b128.
// ---------------------------------------------------------------------------
__global__ __launch_bounds__(256) void attn_pv_kernel(
    const _Float16* __restrict__ P, const _Float16* __restrict__ V,
    _Float16* __restrict__ O, int head) {
    __shared__ __align__(16) _Float16 lA[2][BM * LDT];
    __shared__ __align__(16) _Float16 lB[2][BN * LDT];
    int tid = threadIdx.x, lane = tid & 31, wave = tid >> 5;
    int m0 = blockIdx.y * BM, n0 = blockIdx.x * BN;
    int wm = (wave & 3) * 32, wn = (wave >> 2) * 32;
    int am = tid >> 1, ak = (tid & 1) * 16;
    int bn = tid & 63, bk = (tid >> 6) * 8;
    int cbase = head * HD;
    int kend = m0 + BM; if (kend > T_TOK) kend = T_TOK;
    const _Float16* Ap = P + (size_t)(m0 + am) * T_TOK + ak;
    const _Float16* Bp = V + cbase + n0 + bn;

    v8h a0, a1, hb;
    auto fetch = [&](int k0) {
        a0 = *(const v8h*)(Ap + k0);
        a1 = *(const v8h*)(Ap + k0 + 8);
#pragma unroll
        for (int i = 0; i < 8; ++i) hb[i] = Bp[(size_t)(k0 + bk + i) * HDIM];
    };
    auto stage = [&](int buf) {
        *(v8h*)&lA[buf][am * LDT + ak] = a0;
        *(v8h*)&lA[buf][am * LDT + ak + 8] = a1;
        *(v8h*)&lB[buf][bn * LDT + bk] = hb;
    };

    fetch(0); stage(0);
    v8f c00 = {}, c01 = {}, c10 = {}, c11 = {};
    int buf = 0;
    for (int k0 = 0; k0 < kend; k0 += BK, buf ^= 1) {
        __syncthreads();
        bool nxt = (k0 + BK) < kend;
        if (nxt) fetch(k0 + BK);
        v16h af0 = frag_ld(&lA[buf][wm * LDT], lane);
        v16h af1 = frag_ld(&lA[buf][(wm + 16) * LDT], lane);
        v16h bf0 = frag_ld(&lB[buf][wn * LDT], lane);
        v16h bf1 = frag_ld(&lB[buf][(wn + 16) * LDT], lane);
        c00 = wmma(af0, bf0, c00); c01 = wmma(af0, bf1, c01);
        c10 = wmma(af1, bf0, c10); c11 = wmma(af1, bf1, c11);
        if (nxt) stage(buf ^ 1);
    }
    int col = lane & 15, rb = (lane >> 4) * 8;
#pragma unroll
    for (int r = 0; r < 8; ++r) {
        int mmA = m0 + wm + rb + r, mmB = mmA + 16;
        O[(size_t)mmA * HDIM + cbase + n0 + wn + col]      = (_Float16)c00[r];
        O[(size_t)mmA * HDIM + cbase + n0 + wn + 16 + col] = (_Float16)c01[r];
        O[(size_t)mmB * HDIM + cbase + n0 + wn + col]      = (_Float16)c10[r];
        O[(size_t)mmB * HDIM + cbase + n0 + wn + 16 + col] = (_Float16)c11[r];
    }
}

// ---------------------------------------------------------------------------
// Router: one wave32 per token; lane e owns expert e. Softmax + top-8 via
// shuffle reductions, normalized weights, atomic per-expert counts.
// ---------------------------------------------------------------------------
__global__ void router_kernel(const _Float16* __restrict__ h2, const float* __restrict__ wg,
                              int* __restrict__ topk_id, float* __restrict__ topk_w,
                              int* __restrict__ counts) {
    int t = blockIdx.x, lane = threadIdx.x;
    const _Float16* hr = h2 + (size_t)t * HDIM;
    float dot = 0.f;
    for (int k = 0; k < HDIM; ++k) dot = fmaf((float)hr[k], wg[(size_t)k * NE + lane], dot);
    float mx = dot;
    for (int o = 16; o > 0; o >>= 1) mx = fmaxf(mx, __shfl_xor(mx, o, 32));
    float p = __expf(dot - mx);
    float sum = p;
    for (int o = 16; o > 0; o >>= 1) sum += __shfl_xor(sum, o, 32);
    p /= sum;
    bool sel = false; float ssum = 0.f;
    for (int it = 0; it < TOPK; ++it) {
        float v = sel ? -1.0f : p;
        float mv = v;
        for (int o = 16; o > 0; o >>= 1) mv = fmaxf(mv, __shfl_xor(mv, o, 32));
        unsigned long long b = __ballot(v == mv && !sel);
        int leader = (int)__ffsll((long long)b) - 1;
        if (lane == leader) sel = true;
        ssum += mv;
    }
    unsigned long long sm = __ballot(sel);
    if (sel) {
        int j = (int)__popcll(sm & ((1ull << lane) - 1ull));
        topk_id[t * TOPK + j] = lane;
        topk_w[t * TOPK + j]  = p / ssum;
        atomicAdd(&counts[lane], 1);
    }
}

__global__ void scan_kernel(const int* __restrict__ counts, int* __restrict__ offsets) {
    if (threadIdx.x == 0) { int s = 0; for (int e = 0; e < NE; ++e) { offsets[e] = s; s += counts[e]; } }
}

__global__ void build_kernel(const int* __restrict__ topk_id, const float* __restrict__ topk_w,
                             const int* __restrict__ offsets, int* __restrict__ cursor,
                             int* __restrict__ tok_ids, float* __restrict__ tok_wt) {
    int i = blockIdx.x * blockDim.x + threadIdx.x;
    if (i >= T_TOK * TOPK) return;
    int t = i / TOPK, e = topk_id[i];
    int slot = atomicAdd(&cursor[e], 1);
    tok_ids[offsets[e] + slot] = t;
    tok_wt[offsets[e] + slot]  = topk_w[i];
}

// ---------------------------------------------------------------------------
// Grouped MoE GEMM 1: act[slot,:F] = silu(X@w1[e]) * (X@w3[e]), rows gathered
// by per-expert token lists. Dual-B tile, SwiGLU epilogue, f16 output.
// ---------------------------------------------------------------------------
__global__ __launch_bounds__(256) void moe_gemm1_kernel(
    const _Float16* __restrict__ h2, const float* __restrict__ w1, const float* __restrict__ w3,
    const int* __restrict__ counts, const int* __restrict__ offsets,
    const int* __restrict__ tok_ids, _Float16* __restrict__ act) {
    int e = blockIdx.z, cnt = counts[e], base = offsets[e];
    int m0 = blockIdx.y * BM;
    if (m0 >= cnt) return;
    int n0 = blockIdx.x * BN;
    __shared__ __align__(16) _Float16 lA[2][BM * LDT];
    __shared__ __align__(16) _Float16 lB1[2][BN * LDT];
    __shared__ __align__(16) _Float16 lB3[2][BN * LDT];
    int tid = threadIdx.x, lane = tid & 31, wave = tid >> 5;
    int wm = (wave & 3) * 32, wn = (wave >> 2) * 32;
    int am = tid >> 1, ak = (tid & 1) * 16;
    int bn = tid & 63, bk = (tid >> 6) * 8;
    int slot = m0 + am;
    int tok = (slot < cnt) ? tok_ids[base + slot] : -1;
    const _Float16* Ap = h2 + (tok >= 0 ? (size_t)tok * HDIM : 0) + ak;
    const float* B1p = w1 + (size_t)e * HDIM * FDIM + n0 + bn;
    const float* B3p = w3 + (size_t)e * HDIM * FDIM + n0 + bn;

    v8h a0, a1; float bs1[8], bs3[8];
    auto fetch = [&](int k0) {
        v8h z = {};
        a0 = (tok >= 0) ? *(const v8h*)(Ap + k0)     : z;
        a1 = (tok >= 0) ? *(const v8h*)(Ap + k0 + 8) : z;
        __builtin_prefetch(&B1p[(size_t)(k0 + BK + bk) * FDIM], 0, 1);
        __builtin_prefetch(&B3p[(size_t)(k0 + BK + bk) * FDIM], 0, 1);
#pragma unroll
        for (int i = 0; i < 8; ++i) {
            bs1[i] = B1p[(size_t)(k0 + bk + i) * FDIM];
            bs3[i] = B3p[(size_t)(k0 + bk + i) * FDIM];
        }
    };
    auto stage = [&](int buf) {
        *(v8h*)&lA[buf][am * LDT + ak] = a0;
        *(v8h*)&lA[buf][am * LDT + ak + 8] = a1;
        v8h h1v, h3v;
#pragma unroll
        for (int i = 0; i < 8; ++i) { h1v[i] = (_Float16)bs1[i]; h3v[i] = (_Float16)bs3[i]; }
        *(v8h*)&lB1[buf][bn * LDT + bk] = h1v;
        *(v8h*)&lB3[buf][bn * LDT + bk] = h3v;
    };

    fetch(0); stage(0);
    v8f g00 = {}, g01 = {}, g10 = {}, g11 = {};
    v8f u00 = {}, u01 = {}, u10 = {}, u11 = {};
    int buf = 0;
    for (int k0 = 0; k0 < HDIM; k0 += BK, buf ^= 1) {
        __syncthreads();
        bool nxt = (k0 + BK) < HDIM;
        if (nxt) fetch(k0 + BK);
        v16h af0 = frag_ld(&lA[buf][wm * LDT], lane);
        v16h af1 = frag_ld(&lA[buf][(wm + 16) * LDT], lane);
        v16h b10 = frag_ld(&lB1[buf][wn * LDT], lane);
        v16h b11 = frag_ld(&lB1[buf][(wn + 16) * LDT], lane);
        v16h b30 = frag_ld(&lB3[buf][wn * LDT], lane);
        v16h b31 = frag_ld(&lB3[buf][(wn + 16) * LDT], lane);
        g00 = wmma(af0, b10, g00); g01 = wmma(af0, b11, g01);
        g10 = wmma(af1, b10, g10); g11 = wmma(af1, b11, g11);
        u00 = wmma(af0, b30, u00); u01 = wmma(af0, b31, u01);
        u10 = wmma(af1, b30, u10); u11 = wmma(af1, b31, u11);
        if (nxt) stage(buf ^ 1);
    }
    int col = lane & 15, rb = (lane >> 4) * 8;
#pragma unroll
    for (int r = 0; r < 8; ++r) {
        int sA = m0 + wm + rb + r, sB = sA + 16;
        if (sA < cnt) {
            size_t rowb = (size_t)(base + sA) * FDIM;
            act[rowb + n0 + wn + col]      = (_Float16)(siluf(g00[r]) * u00[r]);
            act[rowb + n0 + wn + 16 + col] = (_Float16)(siluf(g01[r]) * u01[r]);
        }
        if (sB < cnt) {
            size_t rowb = (size_t)(base + sB) * FDIM;
            act[rowb + n0 + wn + col]      = (_Float16)(siluf(g10[r]) * u10[r]);
            act[rowb + n0 + wn + 16 + col] = (_Float16)(siluf(g11[r]) * u11[r]);
        }
    }
}

// ---------------------------------------------------------------------------
// Grouped MoE GEMM 2: out[token,:] += wt * (act @ w2[e]), scatter via f32 atomics
// ---------------------------------------------------------------------------
__global__ __launch_bounds__(256) void moe_gemm2_kernel(
    const _Float16* __restrict__ act, const float* __restrict__ w2,
    const int* __restrict__ counts, const int* __restrict__ offsets,
    const int* __restrict__ tok_ids, const float* __restrict__ tok_wt,
    float* __restrict__ out) {
    int e = blockIdx.z, cnt = counts[e], base = offsets[e];
    int m0 = blockIdx.y * BM;
    if (m0 >= cnt) return;
    int n0 = blockIdx.x * BN;
    __shared__ __align__(16) _Float16 lA[2][BM * LDT];
    __shared__ __align__(16) _Float16 lB[2][BN * LDT];
    int tid = threadIdx.x, lane = tid & 31, wave = tid >> 5;
    int wm = (wave & 3) * 32, wn = (wave >> 2) * 32;
    int am = tid >> 1, ak = (tid & 1) * 16;
    int bn = tid & 63, bk = (tid >> 6) * 8;
    int slot = m0 + am;
    const _Float16* Ap = act + (size_t)(base + (slot < cnt ? slot : 0)) * FDIM + ak;
    const float* Bp = w2 + (size_t)e * FDIM * HDIM + n0 + bn;
    bool arow = (slot < cnt);

    v8h a0, a1; float bs[8];
    auto fetch = [&](int k0) {
        v8h z = {};
        a0 = arow ? *(const v8h*)(Ap + k0)     : z;
        a1 = arow ? *(const v8h*)(Ap + k0 + 8) : z;
        __builtin_prefetch(&Bp[(size_t)(k0 + BK + bk) * HDIM], 0, 1);
#pragma unroll
        for (int i = 0; i < 8; ++i) bs[i] = Bp[(size_t)(k0 + bk + i) * HDIM];
    };
    auto stage = [&](int buf) {
        *(v8h*)&lA[buf][am * LDT + ak] = a0;
        *(v8h*)&lA[buf][am * LDT + ak + 8] = a1;
        v8h hb;
#pragma unroll
        for (int i = 0; i < 8; ++i) hb[i] = (_Float16)bs[i];
        *(v8h*)&lB[buf][bn * LDT + bk] = hb;
    };

    fetch(0); stage(0);
    v8f c00 = {}, c01 = {}, c10 = {}, c11 = {};
    int buf = 0;
    for (int k0 = 0; k0 < FDIM; k0 += BK, buf ^= 1) {
        __syncthreads();
        bool nxt = (k0 + BK) < FDIM;
        if (nxt) fetch(k0 + BK);
        v16h af0 = frag_ld(&lA[buf][wm * LDT], lane);
        v16h af1 = frag_ld(&lA[buf][(wm + 16) * LDT], lane);
        v16h bf0 = frag_ld(&lB[buf][wn * LDT], lane);
        v16h bf1 = frag_ld(&lB[buf][(wn + 16) * LDT], lane);
        c00 = wmma(af0, bf0, c00); c01 = wmma(af0, bf1, c01);
        c10 = wmma(af1, bf0, c10); c11 = wmma(af1, bf1, c11);
        if (nxt) stage(buf ^ 1);
    }
    int col = lane & 15, rb = (lane >> 4) * 8;
#pragma unroll
    for (int r = 0; r < 8; ++r) {
        int sA = m0 + wm + rb + r, sB = sA + 16;
        if (sA < cnt) {
            int t = tok_ids[base + sA]; float w = tok_wt[base + sA];
            atomicAdd(&out[(size_t)t * HDIM + n0 + wn + col],      w * c00[r]);
            atomicAdd(&out[(size_t)t * HDIM + n0 + wn + 16 + col], w * c01[r]);
        }
        if (sB < cnt) {
            int t = tok_ids[base + sB]; float w = tok_wt[base + sB];
            atomicAdd(&out[(size_t)t * HDIM + n0 + wn + col],      w * c10[r]);
            atomicAdd(&out[(size_t)t * HDIM + n0 + wn + 16 + col], w * c11[r]);
        }
    }
}

// ---------------------------------------------------------------------------
extern "C" void kernel_launch(void* const* d_in, const int* in_sizes, int n_in,
                              void* d_out, int out_size, void* d_ws, size_t ws_size,
                              hipStream_t stream) {
    (void)in_sizes; (void)n_in; (void)out_size; (void)ws_size;
    const float* x   = (const float*)d_in[0];
    const float* ln1 = (const float*)d_in[1];
    const float* ln2 = (const float*)d_in[2];
    const float* wq  = (const float*)d_in[3];
    const float* wk  = (const float*)d_in[4];
    const float* wv  = (const float*)d_in[5];
    const float* wo  = (const float*)d_in[6];
    const float* wg  = (const float*)d_in[7];
    const float* w1  = (const float*)d_in[8];
    const float* w3  = (const float*)d_in[9];
    const float* w2  = (const float*)d_in[10];
    float* out = (float*)d_out;

    char* ws = (char*)d_ws;
    size_t off = 0;
    auto alloc = [&](size_t bytes) -> void* {
        void* p = ws + off;
        off += (bytes + 255) & ~(size_t)255;
        return p;
    };
    _Float16* h1    = (_Float16*)alloc((size_t)T_TOK * HDIM * 2);
    _Float16* q     = (_Float16*)alloc((size_t)T_TOK * HDIM * 2);
    _Float16* k     = (_Float16*)alloc((size_t)T_TOK * HDIM * 2);
    _Float16* v     = (_Float16*)alloc((size_t)T_TOK * HDIM * 2);
    float*    S     = (float*)   alloc((size_t)T_TOK * T_TOK * 4);   // one head at a time
    _Float16* P     = (_Float16*)alloc((size_t)T_TOK * T_TOK * 2);
    _Float16* attn  = (_Float16*)alloc((size_t)T_TOK * HDIM * 2);
    float*    resid = (float*)   alloc((size_t)T_TOK * HDIM * 4);
    _Float16* h2    = (_Float16*)alloc((size_t)T_TOK * HDIM * 2);
    int*      tkid  = (int*)     alloc((size_t)T_TOK * TOPK * 4);
    float*    tkw   = (float*)   alloc((size_t)T_TOK * TOPK * 4);
    int*      ctrl  = (int*)     alloc(96 * 4);                      // counts|offsets|cursor
    int*      ids   = (int*)     alloc((size_t)T_TOK * TOPK * 4);
    float*    wts   = (float*)   alloc((size_t)T_TOK * TOPK * 4);
    _Float16* act   = (_Float16*)alloc((size_t)T_TOK * TOPK * FDIM * 2);
    int* counts = ctrl, *offsets = ctrl + 32, *cursor = ctrl + 64;

    hipMemsetAsync(ctrl, 0, 96 * 4, stream);

    // ---- attention block ----
    rmsnorm_kernel<<<T_TOK, 256, 0, stream>>>(x, ln1, h1);
    dim3 gq(HDIM / BN, T_TOK / BM);
    gemm_aw_kernel<0><<<gq, 256, 0, stream>>>(h1, wq, HDIM, HDIM, q, nullptr, nullptr, nullptr);
    gemm_aw_kernel<0><<<gq, 256, 0, stream>>>(h1, wk, HDIM, HDIM, k, nullptr, nullptr, nullptr);
    gemm_aw_kernel<0><<<gq, 256, 0, stream>>>(h1, wv, HDIM, HDIM, v, nullptr, nullptr, nullptr);
    for (int h = 0; h < NHEAD; ++h) {
        attn_score_kernel<<<dim3(T_TOK / BN, T_TOK / BM), 256, 0, stream>>>(q, k, S, h);
        softmax_kernel<<<T_TOK, 256, 0, stream>>>(S, P);
        attn_pv_kernel<<<dim3(HD / BN, T_TOK / BM), 256, 0, stream>>>(P, v, attn, h);
    }
    // O-projection + residual; writes resid (MoE input) and initializes d_out
    gemm_aw_kernel<1><<<gq, 256, 0, stream>>>(attn, wo, HDIM, HDIM, nullptr, resid, x, out);

    // ---- MoE block ----
    rmsnorm_kernel<<<T_TOK, 256, 0, stream>>>(resid, ln2, h2);
    router_kernel<<<T_TOK, 32, 0, stream>>>(h2, wg, tkid, tkw, counts);
    scan_kernel<<<1, 32, 0, stream>>>(counts, offsets);
    build_kernel<<<(T_TOK * TOPK + 255) / 256, 256, 0, stream>>>(tkid, tkw, offsets, cursor, ids, wts);
    moe_gemm1_kernel<<<dim3(FDIM / BN, T_TOK / BM, NE), 256, 0, stream>>>(h2, w1, w3, counts, offsets, ids, act);
    moe_gemm2_kernel<<<dim3(HDIM / BN, T_TOK / BM, NE), 256, 0, stream>>>(act, w2, counts, offsets, ids, wts, out);
}
